// SelfAttentionModule_6004364280161
// MI455X (gfx1250) — matrix operand order repaired and checked by reading, not compiled, not run on hardware
//
#include <hip/hip_runtime.h>

typedef __attribute__((ext_vector_type(8)))  __bf16 v8bf;
typedef __attribute__((ext_vector_type(16))) __bf16 v16bf;
typedef __attribute__((ext_vector_type(8)))  float  v8f;
typedef unsigned short u16;

#define DIM 1024
#define NHEAD 16
#define HDIM 64
#define BATCH 4
#define SEQ 2048
#define ROWS (BATCH * SEQ)          // 8192

// ---------------------------------------------------------------------------
// TDM availability (device pass only; host pass falls back automatically)
// ---------------------------------------------------------------------------
#ifdef __has_builtin
#if __has_builtin(__builtin_amdgcn_tensor_load_to_lds) && \
    __has_builtin(__builtin_amdgcn_s_wait_tensorcnt)
#define USE_TDM 1
#endif
#endif
#ifndef USE_TDM
#define USE_TDM 0
#endif

__device__ __forceinline__ u16 f2bf(float f) {
    union { float f; unsigned u; } v; v.f = f;
    return (u16)((v.u + 0x7fffu + ((v.u >> 16) & 1u)) >> 16);
}

__device__ __forceinline__ v8f vzero8() {
    v8f z;
#pragma unroll
    for (int t = 0; t < 8; ++t) z[t] = 0.0f;
    return z;
}

#if USE_TDM
typedef __attribute__((ext_vector_type(4))) unsigned int u32x4;
typedef __attribute__((ext_vector_type(8))) int          i32x8;
typedef __attribute__((ext_vector_type(4))) int          i32x4;

// TDM 2D tile load: 128 rows x 32 bf16, row stride = row_stride_elems,
// LDS destination padded by hardware: 16 DWORDs (32 u16) + 4 DWORDs pad
// -> LDS row stride = 40 u16 = 80B (matches the As/Bs static layout).
// D# bitfields per CDNA5 ISA.
__device__ __forceinline__ void tdm_load_tile_128x32(const u16* gsrc,
                                                     unsigned lds_byte_off,
                                                     int row_stride_elems) {
    unsigned long long ga = (unsigned long long)(const void*)gsrc;
    u32x4 g0;
    g0.x = 1u;                                        // count=1, user mode
    g0.y = lds_byte_off;                              // lds_addr
    g0.z = (unsigned)ga;                              // global_addr[31:0]
    g0.w = ((unsigned)(ga >> 32) & 0x1FFFFFFu) | (2u << 30);  // addr hi | type=2
    unsigned tdim0 = (unsigned)row_stride_elems;      // tensor dim0 (no OOB here)
    unsigned tdim1 = 1u << 20;                        // large: no OOB
    unsigned long long st0 = (unsigned long long)row_stride_elems;
    i32x8 g1;
    g1[0] = (int)((1u << 16)        // data_size = 1 -> 2 bytes
                | (1u << 20)        // pad_enable
                | (3u << 22)        // pad_interval code 3 -> every 16 DWORDs
                | (3u << 25));      // pad_amount  code 3 -> 4 DWORDs
    g1[1] = (int)((tdim0 & 0xFFFFu) << 16);                         // tdim0 lo
    g1[2] = (int)(((tdim0 >> 16) & 0xFFFFu) | ((tdim1 & 0xFFFFu) << 16));
    g1[3] = (int)(((tdim1 >> 16) & 0xFFFFu) | (32u << 16));         // tile_dim0=32
    g1[4] = (int)128u;                                              // tile_dim1=128
    g1[5] = (int)(unsigned)st0;                                     // stride0 lo
    g1[6] = (int)(unsigned)((st0 >> 32) & 0xFFFFu);                 // stride0 hi
    g1[7] = 0;
    i32x4 z4; z4[0] = 0; z4[1] = 0; z4[2] = 0; z4[3] = 0;
#if defined(__clang_major__) && __clang_major__ >= 23
    i32x8 z8;
#pragma unroll
    for (int t = 0; t < 8; ++t) z8[t] = 0;
    __builtin_amdgcn_tensor_load_to_lds(g0, g1, z4, z4, z8, 0);
#else
    __builtin_amdgcn_tensor_load_to_lds(g0, g1, z4, z4, 0);
#endif
}

__device__ __forceinline__ unsigned lds_off_of(const void* p) {
    // generic LDS address = {aperture_hi32, lds_offset}: low 32 bits = offset
    return (unsigned)(unsigned long long)p;
}
#endif // USE_TDM

// ---------------------------------------------------------------------------
// fp32 -> bf16 elementwise
// ---------------------------------------------------------------------------
__global__ void f32_to_bf16_kernel(const float* __restrict__ in, u16* __restrict__ out, int n) {
    int i = blockIdx.x * blockDim.x + threadIdx.x;
    if (i < n) out[i] = f2bf(in[i]);
}

// ---------------------------------------------------------------------------
// Tiled bf16 GEMM: C[M,N] = A[M,K] @ Bw[N,K]^T
// 256 threads = 8 waves; block tile 128x128; wave tile 64x32; K-step 32.
// Double-buffered LDS; stage t+1 moved by TDM (or global loads) during the
// WMMAs of stage t; one barrier per K-step.
// MODE 0: scatter qkv into head-major q/k/v bf16 buffers (q pre-scaled 1/8)
// MODE 1: fp32 output + bias (final projection)
// ---------------------------------------------------------------------------
template <int MODE>
__global__ __launch_bounds__(256) void gemm_bf16_kernel(
    const u16* __restrict__ A, const u16* __restrict__ Bw,
    int M, int N, int K,
    u16* __restrict__ qb, u16* __restrict__ kb, u16* __restrict__ vb,
    const float* __restrict__ bias, float* __restrict__ Cout)
{
    __shared__ u16 As[2][128][40];   // padded stride 40 u16 = 80B
    __shared__ u16 Bs[2][128][40];

    const int tid  = threadIdx.x;
    const int wave = tid >> 5;
    const int lane = tid & 31;
    const int lh   = lane >> 4;
    const int ln   = lane & 15;
    const int wm   = (wave >> 2) * 64;
    const int wn   = (wave & 3) * 32;
    const int m0   = blockIdx.y * 128;
    const int n0   = blockIdx.x * 128;
    const int NSTAGE = K >> 5;

    v8f acc[4][2];
#pragma unroll
    for (int i = 0; i < 4; ++i)
#pragma unroll
        for (int j = 0; j < 2; ++j) acc[i][j] = vzero8();

#if USE_TDM
    if (wave == 0) {
        tdm_load_tile_128x32(&A[(size_t)m0 * K], lds_off_of(&As[0][0][0]), K);
        tdm_load_tile_128x32(&Bw[(size_t)n0 * K], lds_off_of(&Bs[0][0][0]), K);
        __builtin_amdgcn_s_wait_tensorcnt(0);
    }
    __syncthreads();
#else
    {   // manual prologue stage 0
#pragma unroll
        for (int c = 0; c < 2; ++c) {
            int chunk = tid * 2 + c;
            int r  = chunk >> 2;
            int kc = (chunk & 3) * 8;
            *(uint4*)&As[0][r][kc] = *(const uint4*)&A[(size_t)(m0 + r) * K + kc];
            *(uint4*)&Bs[0][r][kc] = *(const uint4*)&Bw[(size_t)(n0 + r) * K + kc];
        }
    }
    __syncthreads();
#endif

    for (int kt = 0; kt < NSTAGE; ++kt) {
        const int cur = kt & 1, nxt = cur ^ 1;
        const bool have_next = (kt + 1 < NSTAGE);

#if USE_TDM
        if (wave == 0 && have_next) {
            int k1 = (kt + 1) << 5;
            tdm_load_tile_128x32(&A[(size_t)m0 * K + k1], lds_off_of(&As[nxt][0][0]), K);
            tdm_load_tile_128x32(&Bw[(size_t)n0 * K + k1], lds_off_of(&Bs[nxt][0][0]), K);
        }
#else
        uint4 apre[2], bpre[2];
        int pr[2], pk[2];
        if (have_next) {
            int k1 = (kt + 1) << 5;
#pragma unroll
            for (int c = 0; c < 2; ++c) {
                int chunk = tid * 2 + c;
                pr[c] = chunk >> 2;
                pk[c] = (chunk & 3) * 8;
                apre[c] = *(const uint4*)&A[(size_t)(m0 + pr[c]) * K + k1 + pk[c]];
                bpre[c] = *(const uint4*)&Bw[(size_t)(n0 + pr[c]) * K + k1 + pk[c]];
            }
            if (kt + 2 < NSTAGE) {
                int k2 = (kt + 2) << 5;
                __builtin_prefetch(&A[(size_t)(m0 + (tid >> 1)) * K + k2], 0, 0);
                __builtin_prefetch(&Bw[(size_t)(n0 + (tid >> 1)) * K + k2], 0, 0);
            }
        }
#endif

        // ---- compute stage `cur`: 8 WMMAs per wave ----
        v16bf afrag[4];
        v16bf bfrag[2];
#pragma unroll
        for (int i = 0; i < 4; ++i) {
            v8bf lo = *(const v8bf*)&As[cur][wm + 16 * i + ln][lh * 8];
            v8bf hi = *(const v8bf*)&As[cur][wm + 16 * i + ln][16 + lh * 8];
#pragma unroll
            for (int t = 0; t < 8; ++t) { afrag[i][t] = lo[t]; afrag[i][t + 8] = hi[t]; }
        }
#pragma unroll
        for (int j = 0; j < 2; ++j) {
            v8bf lo = *(const v8bf*)&Bs[cur][wn + 16 * j + ln][lh * 8];
            v8bf hi = *(const v8bf*)&Bs[cur][wn + 16 * j + ln][16 + lh * 8];
#pragma unroll
            for (int t = 0; t < 8; ++t) { bfrag[j][t] = lo[t]; bfrag[j][t + 8] = hi[t]; }
        }
#pragma unroll
        for (int i = 0; i < 4; ++i)
#pragma unroll
            for (int j = 0; j < 2; ++j)
                acc[i][j] = __builtin_amdgcn_wmma_f32_16x16x32_bf16(
                    false, afrag[i], false, bfrag[j], (short)0, acc[i][j], false, false);

#if USE_TDM
        if (wave == 0 && have_next) __builtin_amdgcn_s_wait_tensorcnt(0);
#else
        if (have_next) {
#pragma unroll
            for (int c = 0; c < 2; ++c) {
                *(uint4*)&As[nxt][pr[c]][pk[c]] = apre[c];
                *(uint4*)&Bs[nxt][pr[c]][pk[c]] = bpre[c];
            }
        }
#endif
        __syncthreads();
    }

    // epilogue: C/D layout -> row = r + 8*lh, col = ln within each 16x16 block
#pragma unroll
    for (int i = 0; i < 4; ++i)
#pragma unroll
        for (int j = 0; j < 2; ++j)
#pragma unroll
            for (int r = 0; r < 8; ++r) {
                int gm = m0 + wm + 16 * i + r + 8 * lh;
                int gc = n0 + wn + 16 * j + ln;
                float v = acc[i][j][r];
                if (MODE == 0) {
                    int part = gc >> 10, rem = gc & 1023;
                    int h = rem >> 6, d = rem & 63;
                    int b = gm >> 11, nr = gm & 2047;
                    if (part == 0) v *= 0.125f;   // fold softmax scale into Q
                    size_t idx = (((size_t)(b * NHEAD + h)) * SEQ + nr) * HDIM + d;
                    u16* dst = (part == 0) ? qb : (part == 1) ? kb : vb;
                    dst[idx] = f2bf(v);
                } else {
                    Cout[(size_t)gm * N + gc] = v + bias[gc];
                }
            }
}

// ---------------------------------------------------------------------------
// Flash attention: one workgroup (4 waves) handles 64 q-rows of one (b,h).
// Per wave: 16 q-rows; online softmax over KV tiles of 32 rows.
// Q is pre-scaled by 1/sqrt(HDIM) in the QKV epilogue.
// ---------------------------------------------------------------------------
__global__ __launch_bounds__(128) void attn_kernel(
    const u16* __restrict__ qb, const u16* __restrict__ kb,
    const u16* __restrict__ vb, u16* __restrict__ outb)
{
    __shared__ u16 Ks[32][72];        // K tile row-major, padded
    __shared__ u16 Vt[64][40];        // V tile transposed [d][kv], padded
    __shared__ u16 Ps[4][16][40];     // per-wave P (16x32), padded

    const int tid  = threadIdx.x;
    const int wave = tid >> 5;
    const int lane = tid & 31;
    const int lh   = lane >> 4;
    const int ln   = lane & 15;
    const int bh   = blockIdx.x >> 5;       // b*16 + h
    const int qt   = blockIdx.x & 31;
    const int qbase = qt * 64 + wave * 16;

    const u16* Q  = qb + (size_t)bh * SEQ * HDIM;
    const u16* Kp = kb + (size_t)bh * SEQ * HDIM;
    const u16* Vp = vb + (size_t)bh * SEQ * HDIM;

    // Q fragments (K=64 split into two 32-chunks), loaded straight from global
    v16bf qf[2];
#pragma unroll
    for (int c = 0; c < 2; ++c) {
        const u16* row = Q + (size_t)(qbase + ln) * HDIM + c * 32;
        v8bf lo = *(const v8bf*)&row[lh * 8];
        v8bf hi = *(const v8bf*)&row[16 + lh * 8];
#pragma unroll
        for (int t = 0; t < 8; ++t) { qf[c][t] = lo[t]; qf[c][t + 8] = hi[t]; }
    }

    float mrow[8], lrow[8], alpha[8];
#pragma unroll
    for (int r = 0; r < 8; ++r) { mrow[r] = -1e30f; lrow[r] = 0.0f; }
    v8f Oacc[4];
#pragma unroll
    for (int g = 0; g < 4; ++g) Oacc[g] = vzero8();

    for (int j0 = 0; j0 < SEQ; j0 += 32) {
        // stage K (row-major) and V (transposed) tiles
#pragma unroll
        for (int c = 0; c < 2; ++c) {
            int chunk = tid * 2 + c;           // 0..255
            int r   = chunk >> 3;
            int col = (chunk & 7) * 8;
            *(uint4*)&Ks[r][col] = *(const uint4*)&Kp[(size_t)(j0 + r) * HDIM + col];
            uint4 vv = *(const uint4*)&Vp[(size_t)(j0 + r) * HDIM + col];
            const u16* pe = (const u16*)&vv;
#pragma unroll
            for (int e = 0; e < 8; ++e) Vt[col + e][r] = pe[e];
        }
        if (j0 + 32 < SEQ) {
            __builtin_prefetch(&Kp[(size_t)(j0 + 32 + (tid >> 2)) * HDIM], 0, 0);
            __builtin_prefetch(&Vp[(size_t)(j0 + 32 + (tid >> 2)) * HDIM], 0, 0);
        }
        __syncthreads();

        // S = Q @ K^T  (16x32), two 16-col groups x two 32-deep d-chunks
        v8f sacc[2];
        sacc[0] = vzero8(); sacc[1] = vzero8();
#pragma unroll
        for (int cg = 0; cg < 2; ++cg) {
#pragma unroll
            for (int c = 0; c < 2; ++c) {
                v16bf bf;
                const u16* row = &Ks[cg * 16 + ln][c * 32 + lh * 8];
                v8bf lo = *(const v8bf*)&row[0];
                v8bf hi = *(const v8bf*)&row[16];
#pragma unroll
                for (int t = 0; t < 8; ++t) { bf[t] = lo[t]; bf[t + 8] = hi[t]; }
                sacc[cg] = __builtin_amdgcn_wmma_f32_16x16x32_bf16(
                    false, qf[c], false, bf, (short)0, sacc[cg], false, false);
            }
        }

        // online softmax (row r lives in half-wave lh; 2 cols per lane per row)
#pragma unroll
        for (int r = 0; r < 8; ++r) {
            float s0 = sacc[0][r];
            float s1 = sacc[1][r];
            float t = fmaxf(s0, s1);
#pragma unroll
            for (int m = 1; m < 16; m <<= 1) t = fmaxf(t, __shfl_xor(t, m, 32));
            float mnew = fmaxf(mrow[r], t);
            float a  = __expf(mrow[r] - mnew);
            float p0 = __expf(s0 - mnew);
            float p1 = __expf(s1 - mnew);
            float ps = p0 + p1;
#pragma unroll
            for (int m = 1; m < 16; m <<= 1) ps += __shfl_xor(ps, m, 32);
            lrow[r] = lrow[r] * a + ps;
            mrow[r] = mnew;
            alpha[r] = a;
            Ps[wave][r + 8 * lh][ln]      = f2bf(p0);   // same-wave LDS: in-order
            Ps[wave][r + 8 * lh][16 + ln] = f2bf(p1);
        }
#pragma unroll
        for (int g = 0; g < 4; ++g)
#pragma unroll
            for (int r = 0; r < 8; ++r) Oacc[g][r] *= alpha[r];

        // P fragment (A-layout) from per-wave LDS transpose buffer
        v16bf pf;
        {
            const u16* row = &Ps[wave][ln][0];
            v8bf lo = *(const v8bf*)&row[lh * 8];
            v8bf hi = *(const v8bf*)&row[16 + lh * 8];
#pragma unroll
            for (int t = 0; t < 8; ++t) { pf[t] = lo[t]; pf[t + 8] = hi[t]; }
        }

        // O += P(16x32) @ V(32x64): 4 d-col groups
#pragma unroll
        for (int g = 0; g < 4; ++g) {
            v16bf vf;
            const u16* row = &Vt[g * 16 + ln][lh * 8];
            v8bf lo = *(const v8bf*)&row[0];
            v8bf hi = *(const v8bf*)&row[16];
#pragma unroll
            for (int t = 0; t < 8; ++t) { vf[t] = lo[t]; vf[t + 8] = hi[t]; }
            Oacc[g] = __builtin_amdgcn_wmma_f32_16x16x32_bf16(
                false, pf, false, vf, (short)0, Oacc[g], false, false);
        }
        __syncthreads();
    }

    // epilogue: normalize and write head-interleaved bf16
    const int h = bh & (NHEAD - 1);
    const int b = bh >> 4;
#pragma unroll
    for (int g = 0; g < 4; ++g)
#pragma unroll
        for (int r = 0; r < 8; ++r) {
            int qr  = qbase + r + 8 * lh;
            int col = h * HDIM + g * 16 + ln;
            float v = Oacc[g][r] / lrow[r];
            outb[((size_t)(b * SEQ + qr)) * DIM + col] = f2bf(v);
        }
}

// ---------------------------------------------------------------------------
// Launch
// ---------------------------------------------------------------------------
extern "C" void kernel_launch(void* const* d_in, const int* in_sizes, int n_in,
                              void* d_out, int out_size, void* d_ws, size_t ws_size,
                              hipStream_t stream) {
    const float* x     = (const float*)d_in[0];   // [4,2048,1024]
    const float* wqkv  = (const float*)d_in[1];   // [3072,1024]
    const float* wproj = (const float*)d_in[2];   // [1024,1024]
    const float* bproj = (const float*)d_in[3];   // [1024]

    char* ws = (char*)d_ws;
    const size_t NX    = (size_t)ROWS * DIM;          // 8,388,608
    const size_t NWQKV = (size_t)3 * DIM * DIM;       // 3,145,728
    const size_t NWPRJ = (size_t)DIM * DIM;           // 1,048,576
    const size_t NHQKV = (size_t)ROWS * DIM;

    u16* xb     = (u16*)ws;                                    ws += NX * 2;
    u16* wqkvb  = (u16*)ws;                                    ws += NWQKV * 2;
    u16* wprojb = (u16*)ws;                                    ws += NWPRJ * 2;
    u16* qbuf   = (u16*)ws;                                    ws += NHQKV * 2;
    u16* kbuf   = (u16*)ws;                                    ws += NHQKV * 2;
    u16* vbuf   = (u16*)ws;                                    ws += NHQKV * 2;
    u16* attb   = (u16*)ws;                                    ws += NX * 2;

    f32_to_bf16_kernel<<<(int)((NX    + 255) / 256), 256, 0, stream>>>(x,     xb,     (int)NX);
    f32_to_bf16_kernel<<<(int)((NWQKV + 255) / 256), 256, 0, stream>>>(wqkv,  wqkvb,  (int)NWQKV);
    f32_to_bf16_kernel<<<(int)((NWPRJ + 255) / 256), 256, 0, stream>>>(wproj, wprojb, (int)NWPRJ);

    // QKV GEMM: [8192,1024] x [3072,1024]^T, scatter into head-major q/k/v
    gemm_bf16_kernel<0><<<dim3(3 * DIM / 128, ROWS / 128), 256, 0, stream>>>(
        xb, wqkvb, ROWS, 3 * DIM, DIM, qbuf, kbuf, vbuf, nullptr, nullptr);

    // Flash attention: 64 (b,h) pairs x 32 q-tiles of 64 rows
    attn_kernel<<<BATCH * NHEAD * (SEQ / 64), 128, 0, stream>>>(qbuf, kbuf, vbuf, attb);

    // Projection: [8192,1024] x [1024,1024]^T + bias -> fp32 d_out
    gemm_bf16_kernel<1><<<dim3(DIM / 128, ROWS / 128), 256, 0, stream>>>(
        attb, wprojb, ROWS, DIM, DIM, nullptr, nullptr, nullptr, bproj, (float*)d_out);
}
